// NaiveLSTM_84361747628164
// MI455X (gfx1250) — compile-verified, matrix operand order checked by reading, not run on hardware
//
#include <hip/hip_runtime.h>
#include <hip/hip_bf16.h>

// ---------------------------------------------------------------------------
// LSTM on MI455X (gfx1250), wave32 + WMMA bf16.
//   B=64, T=512, D=256, H=512, O=256, G=4H=2048
// Phase 0: prep    - bf16 weight repack (per-WG transposed blocks), bias fold
// Phase 1: recur   - persistent kernel, 32 WGs, grid barrier per timestep,
//                    two-buffer pipelined weight fragments + x-slab prefetch
// Phase 2: outgemm - y = hs @ W_y + b_y  (WMMA tiled GEMM, next-chunk prefetch)
// ---------------------------------------------------------------------------

#define LSTM_B 64
#define LSTM_T 512
#define LSTM_D 256
#define LSTM_H 512
#define LSTM_G 2048      // 4*H
#define LSTM_O 256
#define LSTM_K 768       // D + H  (fused [x_t | h] contraction)
#define NWG    32        // persistent workgroups; each owns 16 hidden cols
#define A_STRIDE 776     // 768 + 8 bf16 pad (keeps 16B alignment, breaks bank conflicts)

typedef __attribute__((ext_vector_type(16))) __bf16 v16bf;
typedef __attribute__((ext_vector_type(8)))  float  v8f;

union Frag16 {           // 16 bf16 = 8 VGPRs = two 128-bit loads
  v16bf v;
  uint4 q[2];
};

__device__ __forceinline__ unsigned short f2bf(float f) {
  union { float f; unsigned u; } v; v.f = f;
  unsigned r = v.u + 0x7FFFu + ((v.u >> 16) & 1u);   // round-to-nearest-even
  return (unsigned short)(r >> 16);
}

__device__ __forceinline__ float sigm(float x) {
  return 1.0f / (1.0f + __expf(-x));
}
__device__ __forceinline__ float tanh_fast(float x) {
  return 1.0f - 2.0f / (__expf(2.0f * x) + 1.0f);
}

__device__ __forceinline__ v8f wmma_bf16(const Frag16& a, const Frag16& b, v8f c) {
  // (neg_a, A, neg_b, B, c_mod, C, reuse_a, reuse_b)
  return __builtin_amdgcn_wmma_f32_16x16x32_bf16(false, a.v, false, b.v,
                                                 (short)0, c, false, false);
}

// ---------------------------------------------------------------------------
// Phase 0: weight repack + bias fold + state init
//   Wp  [NWG][64][768] bf16 : per-WG column blocks of [W_x ; U_h],
//                             local col n -> gate g=n/16, hidden col wg*16+n%16
//   Wyp [256][512]     bf16 : W_y transposed (col-major for B fragments)
//   bb  [2048]         f32  : b_ih + b_hh
//   hbuf[64][512]      bf16 : h state ping buffer, init from h0
// ---------------------------------------------------------------------------
__global__ void lstm_prep(const float* __restrict__ W_x, const float* __restrict__ U_h,
                          const float* __restrict__ W_y, const float* __restrict__ b_ih,
                          const float* __restrict__ b_hh, const float* __restrict__ h0,
                          unsigned short* __restrict__ Wp, unsigned short* __restrict__ Wyp,
                          float* __restrict__ bb, unsigned short* __restrict__ hbuf,
                          unsigned int* __restrict__ gsync) {
  unsigned idx = blockIdx.x * blockDim.x + threadIdx.x;
  if (idx < (unsigned)(NWG * 64 * LSTM_K)) {            // 1,572,864
    int wg = idx / (64 * LSTM_K);
    int n  = (idx / LSTM_K) % 64;
    int k  = idx % LSTM_K;
    int g = n >> 4, hc = n & 15;
    int col = g * LSTM_H + wg * 16 + hc;                // column of [W_x;U_h]
    float v = (k < LSTM_D) ? W_x[(size_t)k * LSTM_G + col]
                           : U_h[(size_t)(k - LSTM_D) * LSTM_G + col];
    Wp[idx] = f2bf(v);
  }
  if (idx < (unsigned)(LSTM_O * LSTM_H)) {              // 131,072 : Wyp[n][k] = W_y[k][n]
    int n = idx >> 9, k = idx & 511;
    Wyp[idx] = f2bf(W_y[(size_t)k * LSTM_O + n]);
  }
  if (idx < LSTM_G)               bb[idx]   = b_ih[idx] + b_hh[idx];
  if (idx < LSTM_B * LSTM_H)      hbuf[idx] = f2bf(h0[idx]);
  if (idx == 0)                   *gsync    = 0u;
}

// ---------------------------------------------------------------------------
// Phase 1: persistent recurrence.  Per step, per WG:
//   gates[64 x 64] = A[64 x 768] * Wblk[768 x 64],  A = [x_t | h] in LDS bf16
//   4x4 WMMA tiles, 8 waves, 2 tiles/wave, K-loop of 24 x (16x16x32 bf16)
//   with the weight (B) fragment pipelined via two alternating register
//   buffers (no rotation copy, no WAR-hazard nops).
//   Then fp32 gate math; c persists in LDS; h -> global (bf16) + hs trace.
//   Device-wide barrier (release fence + atomic + acquire fence) each step.
// ---------------------------------------------------------------------------
__global__ void lstm_recur(const float* __restrict__ x, const float* __restrict__ c0,
                           const unsigned short* __restrict__ Wp,
                           const float* __restrict__ bb,
                           unsigned short* __restrict__ hbuf,
                           unsigned short* __restrict__ hs,
                           float* __restrict__ out_hT, float* __restrict__ out_cT,
                           unsigned int* __restrict__ gsync) {
  extern __shared__ unsigned char smem[];
  unsigned short* aS = (unsigned short*)smem;                         // [64][A_STRIDE] bf16
  float* gS = (float*)(smem + LSTM_B * A_STRIDE * 2);                 // [64][64] gates f32
  float* cS = (float*)(smem + LSTM_B * A_STRIDE * 2 + 64 * 64 * 4);   // [64][16] c state f32

  const int tid  = threadIdx.x;
  const int wg   = blockIdx.x;            // 0..31
  const int lane = tid & 31;
  const int wave = tid >> 5;              // 0..7
  const int nt   = wave & 3;              // column tile == gate index
  const int mt0  = (wave >> 2) * 2;       // row tiles mt0, mt0+1
  const int half16 = lane >> 4;           // K-half select per ISA A/B layout
  const int lrow   = lane & 15;

  const unsigned short* Wblk = Wp + (size_t)wg * 64 * LSTM_K;
  const unsigned short* bpBase  = &Wblk[(nt * 16 + lrow) * LSTM_K + half16 * 8];
  // per-wave LDS A-fragment bases (row tiles mt0, mt0+1)
  const unsigned short* apBase0 = &aS[(mt0 * 16 + lrow) * A_STRIDE + half16 * 8];
  const unsigned short* apBase1 = apBase0 + 16 * A_STRIDE;

  // persistent c state for this WG's 16 hidden columns
#pragma unroll
  for (int i = 0; i < 4; ++i) {
    int e = i * 256 + tid;                // e = b*16 + hc
    int b = e >> 4, hc = e & 15;
    cS[e] = c0[b * LSTM_H + wg * 16 + hc];
  }
  const int hcE  = tid & 15;
  const int colE = wg * 16 + hcE;         // this thread's hidden column (elementwise)
  const float bi = bb[0 * LSTM_H + colE];
  const float bf = bb[1 * LSTM_H + colE];
  const float bj = bb[2 * LSTM_H + colE];
  const float bo = bb[3 * LSTM_H + colE];
  __syncthreads();

  for (int t = 0; t < LSTM_T; ++t) {
    // ---- stage A = [x_t (fp32->bf16) | h (bf16)] into LDS ----
#pragma unroll 4
    for (int it = 0; it < 16; ++it) {     // x part: 64x256 via float4
      int b = it * 4 + (tid >> 6);
      int d = (tid & 63) * 4;
      float4 v = *(const float4*)&x[((size_t)b * LSTM_T + t) * LSTM_D + d];
      uint2 pk;
      pk.x = (unsigned)f2bf(v.x) | ((unsigned)f2bf(v.y) << 16);
      pk.y = (unsigned)f2bf(v.z) | ((unsigned)f2bf(v.w) << 16);
      *(uint2*)&aS[b * A_STRIDE + d] = pk;
    }
#pragma unroll 4
    for (int it = 0; it < 16; ++it) {     // h part: 64x512 bf16 via uint4
      int b = it * 4 + (tid >> 6);
      int k = (tid & 63) * 8;
      *(uint4*)&aS[b * A_STRIDE + LSTM_D + k] = *(const uint4*)&hbuf[b * LSTM_H + k];
    }

    // ---- prefetch next timestep's x slab (64KB) behind the K-loop ----
    if (t + 1 < LSTM_T) {
#pragma unroll
      for (int p = 0; p < 2; ++p) {
        int e = p * 256 + tid;            // 512 cachelines of 128B
        int b = e >> 3;
        int d = (e & 7) * 32;             // 32 floats = 128B per line
        __builtin_prefetch(&x[((size_t)b * LSTM_T + (t + 1)) * LSTM_D + d], 0, 3);
      }
    }
    __syncthreads();

    // ---- K-loop: 24 x wmma_f32_16x16x32_bf16, 2 M-tiles per wave;
    //      weight fragment double-buffered across alternating registers ----
    v8f acc0 = {}; v8f acc1 = {};
    Frag16 fbA, fbB;
    fbA.q[0] = *(const uint4*)(bpBase);
    fbA.q[1] = *(const uint4*)(bpBase + 16);
#pragma unroll 2
    for (int ks = 0; ks < 24; ks += 2) {
      // issue weight fragment ks+1 while ks computes
      {
        const unsigned short* bp = bpBase + (ks + 1) * 32;
        fbB.q[0] = *(const uint4*)bp;
        fbB.q[1] = *(const uint4*)(bp + 16);
      }
      Frag16 fa0, fa1;
      const unsigned short* ap0 = apBase0 + ks * 32;
      fa0.q[0] = *(const uint4*)ap0;
      fa0.q[1] = *(const uint4*)(ap0 + 16);
      const unsigned short* ap1 = apBase1 + ks * 32;
      fa1.q[0] = *(const uint4*)ap1;
      fa1.q[1] = *(const uint4*)(ap1 + 16);
      acc0 = wmma_bf16(fa0, fbA, acc0);
      acc1 = wmma_bf16(fa1, fbA, acc1);
      // issue weight fragment ks+2 while ks+1 computes
      if (ks + 2 < 24) {
        const unsigned short* bp = bpBase + (ks + 2) * 32;
        fbA.q[0] = *(const uint4*)bp;
        fbA.q[1] = *(const uint4*)(bp + 16);
      }
      const unsigned short* aq0 = apBase0 + (ks + 1) * 32;
      fa0.q[0] = *(const uint4*)aq0;
      fa0.q[1] = *(const uint4*)(aq0 + 16);
      const unsigned short* aq1 = apBase1 + (ks + 1) * 32;
      fa1.q[0] = *(const uint4*)aq1;
      fa1.q[1] = *(const uint4*)(aq1 + 16);
      acc0 = wmma_bf16(fa0, fbB, acc0);
      acc1 = wmma_bf16(fa1, fbB, acc1);
    }

    // ---- spill gate tiles to LDS (C/D layout: lane=N, m = vgpr + 8*(lane>=16)) ----
#pragma unroll
    for (int r = 0; r < 8; ++r) {
      gS[(mt0 * 16 + r + half16 * 8) * 64 + nt * 16 + lrow]       = acc0[r];
      gS[((mt0 + 1) * 16 + r + half16 * 8) * 64 + nt * 16 + lrow] = acc1[r];
    }
    __syncthreads();

    // ---- elementwise gate math; c stays in LDS ----
#pragma unroll
    for (int i = 0; i < 4; ++i) {
      int e = i * 256 + tid;              // e = b*16 + hc, hc == hcE
      int b = e >> 4;
      float gi = sigm(gS[b * 64 +  0 + hcE] + bi);
      float gf = sigm(gS[b * 64 + 16 + hcE] + bf);
      float gj = tanh_fast(gS[b * 64 + 32 + hcE] + bj);
      float go = sigm(gS[b * 64 + 48 + hcE] + bo);
      float c  = gf * cS[e] + gi * gj;
      cS[e] = c;
      float h = go * tanh_fast(c);
      unsigned short hb = f2bf(h);
      hbuf[b * LSTM_H + colE] = hb;
      hs[((size_t)b * LSTM_T + t) * LSTM_H + colE] = hb;
      if (t == LSTM_T - 1) {
        out_hT[b * LSTM_H + colE] = h;
        out_cT[b * LSTM_H + colE] = c;
      }
    }

    // ---- device-wide step barrier ----
    __builtin_amdgcn_fence(__ATOMIC_RELEASE, "agent");     // flush h writes (global_wb)
    __syncthreads();
    if (tid == 0) {
      __hip_atomic_fetch_add(gsync, 1u, __ATOMIC_RELEASE, __HIP_MEMORY_SCOPE_AGENT);
      unsigned target = (unsigned)(t + 1) * NWG;
      while (__hip_atomic_load(gsync, __ATOMIC_ACQUIRE, __HIP_MEMORY_SCOPE_AGENT) < target)
        __builtin_amdgcn_s_sleep(2);
    }
    __syncthreads();
    __builtin_amdgcn_fence(__ATOMIC_ACQUIRE, "agent");     // invalidate stale h (global_inv)
  }
}

// ---------------------------------------------------------------------------
// Phase 2: y[32768 x 256] = hs[32768 x 512] @ W_y + b_y, bf16 WMMA, 64x64 tiles
// ---------------------------------------------------------------------------
__global__ void lstm_outgemm(const unsigned short* __restrict__ hs,
                             const unsigned short* __restrict__ Wyp,  // [256][512]
                             const float* __restrict__ b_y,
                             float* __restrict__ y) {
  __shared__ unsigned short aS[64 * 136];   // 128 K-chunk + 8 pad
  __shared__ unsigned short bS[64 * 136];
  const int tid  = threadIdx.x;
  const int lane = tid & 31, wave = tid >> 5;
  const int nt = wave & 3, mt0 = (wave >> 2) * 2;
  const int half16 = lane >> 4, lrow = lane & 15;
  const int rb = blockIdx.x >> 2;           // 0..511  (rows of 64)
  const int cb = blockIdx.x & 3;            // 0..3    (cols of 64)

  v8f acc0 = {}; v8f acc1 = {};
  for (int kc = 0; kc < 4; ++kc) {
    int kbase = kc * 128;
#pragma unroll
    for (int it = 0; it < 4; ++it) {        // stage 64x128 of A and B
      int r = it * 16 + (tid >> 4);
      int k = (tid & 15) * 8;
      *(uint4*)&aS[r * 136 + k] = *(const uint4*)&hs[(size_t)(rb * 64 + r) * LSTM_H + kbase + k];
      *(uint4*)&bS[r * 136 + k] = *(const uint4*)&Wyp[(size_t)(cb * 64 + r) * LSTM_H + kbase + k];
    }
    // prefetch next K-chunk (128B lines) while this chunk's WMMAs run
    if (kc < 3) {
      int nk = kbase + 128;
      int r2 = (tid & 127) >> 1;
      int k2 = (tid & 1) * 64;              // 64 bf16 = 128B line
      if (tid < 128)
        __builtin_prefetch(&hs[(size_t)(rb * 64 + r2) * LSTM_H + nk + k2], 0, 3);
      else
        __builtin_prefetch(&Wyp[(size_t)(cb * 64 + r2) * LSTM_H + nk + k2], 0, 3);
    }
    __syncthreads();
#pragma unroll
    for (int ks = 0; ks < 4; ++ks) {
      int kk = ks * 32 + half16 * 8;
      Frag16 fb, fa0, fa1;
      fb.q[0]  = *(const uint4*)&bS[(nt * 16 + lrow) * 136 + kk];
      fb.q[1]  = *(const uint4*)&bS[(nt * 16 + lrow) * 136 + kk + 16];
      fa0.q[0] = *(const uint4*)&aS[(mt0 * 16 + lrow) * 136 + kk];
      fa0.q[1] = *(const uint4*)&aS[(mt0 * 16 + lrow) * 136 + kk + 16];
      acc0 = wmma_bf16(fa0, fb, acc0);
      fa1.q[0] = *(const uint4*)&aS[((mt0 + 1) * 16 + lrow) * 136 + kk];
      fa1.q[1] = *(const uint4*)&aS[((mt0 + 1) * 16 + lrow) * 136 + kk + 16];
      acc1 = wmma_bf16(fa1, fb, acc1);
    }
    __syncthreads();
  }
  int n = cb * 64 + nt * 16 + lrow;
  float bias = b_y[n];
#pragma unroll
  for (int r = 0; r < 8; ++r) {
    int m0 = rb * 64 + mt0 * 16 + r + half16 * 8;
    y[(size_t)m0 * LSTM_O + n] = acc0[r] + bias;
    y[(size_t)(m0 + 16) * LSTM_O + n] = acc1[r] + bias;
  }
}

// ---------------------------------------------------------------------------
// Workspace layout (bytes):
//   Wp   @ 0         : 3,145,728   bf16 [32][64][768]
//   Wyp  @ 3145728   :   262,144   bf16 [256][512]
//   bb   @ 3407872   :     8,192   f32  [2048]
//   hbuf @ 3416064   :    65,536   bf16 [64][512]
//   sync @ 3481600   :        64
//   hs   @ 3481664   : 33,554,432  bf16 [64][512][512]      total ~37 MB
// ---------------------------------------------------------------------------
extern "C" void kernel_launch(void* const* d_in, const int* in_sizes, int n_in,
                              void* d_out, int out_size, void* d_ws, size_t ws_size,
                              hipStream_t stream) {
  (void)in_sizes; (void)n_in; (void)out_size; (void)ws_size;
  const float* x    = (const float*)d_in[0];
  const float* h0   = (const float*)d_in[1];
  const float* c0   = (const float*)d_in[2];
  const float* W_x  = (const float*)d_in[3];
  const float* U_h  = (const float*)d_in[4];
  const float* b_ih = (const float*)d_in[5];
  const float* b_hh = (const float*)d_in[6];
  const float* W_y  = (const float*)d_in[7];
  const float* b_y  = (const float*)d_in[8];
  float* out = (float*)d_out;

  unsigned char* ws = (unsigned char*)d_ws;
  unsigned short* Wp   = (unsigned short*)(ws + 0);
  unsigned short* Wyp  = (unsigned short*)(ws + 3145728);
  float*          bb   = (float*)(ws + 3407872);
  unsigned short* hbuf = (unsigned short*)(ws + 3416064);
  unsigned int*   gs   = (unsigned int*)(ws + 3481600);
  unsigned short* hs   = (unsigned short*)(ws + 3481664);

  float* out_hT = out + (size_t)LSTM_B * LSTM_T * LSTM_O;   // 8,388,608
  float* out_cT = out_hT + LSTM_B * LSTM_H;                 // +32,768

  lstm_prep<<<6144, 256, 0, stream>>>(W_x, U_h, W_y, b_ih, b_hh, h0,
                                      Wp, Wyp, bb, hbuf, gs);

  size_t dynLds = (size_t)LSTM_B * A_STRIDE * 2 + 64 * 64 * 4 + 64 * 16 * 4; // 119,808 B
  (void)hipFuncSetAttribute((const void*)lstm_recur,
                            hipFuncAttributeMaxDynamicSharedMemorySize, (int)dynLds);
  lstm_recur<<<NWG, 256, dynLds, stream>>>(x, c0, Wp, bb, hbuf, hs,
                                           out_hT, out_cT, gs);

  lstm_outgemm<<<2048, 256, 0, stream>>>(hs, Wyp, b_y, out);
}
